// TimeRNNModel_84799834292551
// MI455X (gfx1250) — matrix operand order repaired and checked by reading, not compiled, not run on hardware
//
#include <hip/hip_runtime.h>
#include <hip/hip_bf16.h>
#include <cstdint>
#include <cstddef>

#define T_STEPS 256
#define BATCH   64
#define HID     512
#define G3H     1536
#define REC_WGS 16   // 32 H-columns per workgroup

typedef __attribute__((ext_vector_type(16))) _Float16 v16h;
typedef __attribute__((ext_vector_type(8)))  _Float16 v8h;
typedef __attribute__((ext_vector_type(4)))  _Float16 v4h;
typedef __attribute__((ext_vector_type(8)))  float    v8f;
typedef __attribute__((ext_vector_type(4)))  float    v4f;

__device__ __forceinline__ float sigmoidf_(float x) { return 1.0f / (1.0f + __expf(-x)); }

// ---------------------------------------------------------------------------
// Fragment loaders (wave32 WMMA layouts per CDNA5 ISA 7.12.2)
// ---------------------------------------------------------------------------

// A fragment (16x32 f16) from f16 row-major [*, ld] at tile (m0, k0).
// Lane L: M = L&15 ; K = e + (e>=8 ? 8 : 0) + (L>=16 ? 8 : 0)
// => per lane: halves [kb, kb+8) and [kb+16, kb+24) : two 16B loads, no cvt.
__device__ __forceinline__ v16h load_a_frag_f16(const _Float16* __restrict__ A, int ld, int m0, int k0) {
  const int lane = threadIdx.x & 31;
  const int row  = m0 + (lane & 15);
  const int kb   = k0 + ((lane >> 4) << 3);
  const _Float16* p = A + (size_t)row * ld + kb;
  v8h lo = *(const v8h*)(p);
  v8h hi = *(const v8h*)(p + 16);
  return __builtin_shufflevector(lo, hi, 0, 1, 2, 3, 4, 5, 6, 7, 8, 9, 10, 11, 12, 13, 14, 15);
}

// B fragment (32x16 f16) from fragment-linear packed weights.
// Packed tile = 512 halves; lane L reads halves [L*16, L*16+16) -> one 32B load.
__device__ __forceinline__ v16h load_b_frag(const _Float16* __restrict__ P, int ntiles, int n0, int k0) {
  const int lane = threadIdx.x & 31;
  const _Float16* p = P + (((size_t)(k0 >> 5) * ntiles + (n0 >> 4)) << 9) + (lane << 4);
  return *(const v16h*)p;
}

// ---------------------------------------------------------------------------
// Weight packing: f32 row-major W[N][K] -> fragment-linear f16
// B layout: lane L: N = L&15 ; K = (e&7) + (e>=8 ? 16 : 0) + (L>=16 ? 8 : 0)
// ---------------------------------------------------------------------------
__global__ void pack_w_kernel(const float* __restrict__ W, _Float16* __restrict__ P, int N, int Kd) {
  size_t o = (size_t)blockIdx.x * blockDim.x + threadIdx.x;
  size_t total = (size_t)N * Kd;
  if (o >= total) return;
  int ntiles = N >> 4;
  int tile   = (int)(o >> 9);
  int within = (int)(o & 511);
  int lane   = within >> 4;
  int e      = within & 15;
  int kt     = tile / ntiles;
  int nt     = tile - kt * ntiles;
  int n      = nt * 16 + (lane & 15);
  int k      = kt * 32 + (e & 7) + ((e & 8) ? 16 : 0) + ((lane & 16) ? 8 : 0);
  P[o] = (_Float16)W[(size_t)n * Kd + k];
}

// Elementwise f32 -> f16 (for the initial x activation copy)
__global__ void cvt_f16_kernel(const float* __restrict__ S, _Float16* __restrict__ D, int n4) {
  int i = blockIdx.x * blockDim.x + threadIdx.x;
  if (i >= n4) return;
  v4f v = *(const v4f*)(S + (size_t)i * 4);
  v4h h;
#pragma unroll
  for (int j = 0; j < 4; ++j) h[j] = (_Float16)v[j];
  *(v4h*)(D + (size_t)i * 4) = h;
}

// ---------------------------------------------------------------------------
// Big GEMM: Out[M,N] = Xh[M,512] @ W^T + bias   (input projections)
// Block: 256 thr = 8 waves; block tile 64x128; wave tile 16x64 (4 accum tiles)
// Double-buffered k-loop: loads for k0+32 overlap WMMAs for k0.
// ---------------------------------------------------------------------------
__global__ __launch_bounds__(256) void gemm_bias_kernel(
    const _Float16* __restrict__ Xh, const _Float16* __restrict__ Wp,
    const float* __restrict__ bias, float* __restrict__ Out, int M, int N) {
  const int wave = threadIdx.x >> 5;
  const int lane = threadIdx.x & 31;
  const int m0 = blockIdx.x * 64 + (wave >> 1) * 16;
  const int n0 = blockIdx.y * 128 + (wave & 1) * 64;
  const int ntiles = N >> 4;

  v8f acc[4];
#pragma unroll
  for (int c = 0; c < 4; ++c)
#pragma unroll
    for (int j = 0; j < 8; ++j) acc[c][j] = 0.0f;

  v16h a_cur = load_a_frag_f16(Xh, HID, m0, 0);
  v16h b_cur[4];
#pragma unroll
  for (int c = 0; c < 4; ++c) b_cur[c] = load_b_frag(Wp, ntiles, n0 + c * 16, 0);

  for (int k0 = 0; k0 < HID; k0 += 32) {
    v16h a_nxt = a_cur;
    v16h b_nxt[4];
    const int kn = k0 + 32;
    if (kn < HID) {
      a_nxt = load_a_frag_f16(Xh, HID, m0, kn);
#pragma unroll
      for (int c = 0; c < 4; ++c) b_nxt[c] = load_b_frag(Wp, ntiles, n0 + c * 16, kn);
    }
#pragma unroll
    for (int c = 0; c < 4; ++c)
      acc[c] = __builtin_amdgcn_wmma_f32_16x16x32_f16(false, a_cur, false, b_cur[c], (short)0, acc[c], false, false);
    a_cur = a_nxt;
#pragma unroll
    for (int c = 0; c < 4; ++c) b_cur[c] = (kn < HID) ? b_nxt[c] : b_cur[c];
  }

  const int nl = lane & 15;
  const int mb = m0 + ((lane >> 4) << 3);
#pragma unroll
  for (int c = 0; c < 4; ++c) {
    int col = n0 + c * 16 + nl;
    float bv = bias[col];
#pragma unroll
    for (int v = 0; v < 8; ++v)
      Out[(size_t)(mb + v) * N + col] = acc[c][v] + bv;
  }
}

// ---------------------------------------------------------------------------
// Gated residual mix: v = sigmoid(Yh @ linW^T + b); Out = Y*v + X*(1-v)
// Writes f32 result + f16 copy (A input of next GEMM).
// ---------------------------------------------------------------------------
__global__ __launch_bounds__(256) void gate_mix_kernel(
    const _Float16* __restrict__ Yh, const float* __restrict__ Yf,
    const _Float16* __restrict__ Wp, const float* __restrict__ bias,
    const float* __restrict__ Xres, float* __restrict__ Out,
    _Float16* __restrict__ Outh, int M) {
  const int wave = threadIdx.x >> 5;
  const int lane = threadIdx.x & 31;
  const int m0 = blockIdx.x * 64 + (wave >> 1) * 16;
  const int n0 = blockIdx.y * 128 + (wave & 1) * 64;
  const int ntiles = HID >> 4;

  v8f acc[4];
#pragma unroll
  for (int c = 0; c < 4; ++c)
#pragma unroll
    for (int j = 0; j < 8; ++j) acc[c][j] = 0.0f;

  v16h a_cur = load_a_frag_f16(Yh, HID, m0, 0);
  v16h b_cur[4];
#pragma unroll
  for (int c = 0; c < 4; ++c) b_cur[c] = load_b_frag(Wp, ntiles, n0 + c * 16, 0);

  for (int k0 = 0; k0 < HID; k0 += 32) {
    v16h a_nxt = a_cur;
    v16h b_nxt[4];
    const int kn = k0 + 32;
    if (kn < HID) {
      a_nxt = load_a_frag_f16(Yh, HID, m0, kn);
#pragma unroll
      for (int c = 0; c < 4; ++c) b_nxt[c] = load_b_frag(Wp, ntiles, n0 + c * 16, kn);
    }
#pragma unroll
    for (int c = 0; c < 4; ++c)
      acc[c] = __builtin_amdgcn_wmma_f32_16x16x32_f16(false, a_cur, false, b_cur[c], (short)0, acc[c], false, false);
    a_cur = a_nxt;
#pragma unroll
    for (int c = 0; c < 4; ++c) b_cur[c] = (kn < HID) ? b_nxt[c] : b_cur[c];
  }

  const int nl = lane & 15;
  const int mb = m0 + ((lane >> 4) << 3);
#pragma unroll
  for (int c = 0; c < 4; ++c) {
    int col = n0 + c * 16 + nl;
    float bv = bias[col];
#pragma unroll
    for (int v = 0; v < 8; ++v) {
      size_t idx = (size_t)(mb + v) * HID + col;
      float g = sigmoidf_(acc[c][v] + bv);
      float z = Yf[idx] * g + Xres[idx] * (1.0f - g);
      Out[idx] = z;
      Outh[idx] = (_Float16)z;
    }
  }
}

// ---------------------------------------------------------------------------
// Init: zero h double-buffer and barrier words (ws is poisoned by harness)
// ---------------------------------------------------------------------------
__global__ void init_state_kernel(float* __restrict__ hbuf, unsigned* __restrict__ bar) {
  int i = blockIdx.x * blockDim.x + threadIdx.x;
  const int n = 2 * BATCH * HID;
  if (i < n) hbuf[i] = 0.0f;
  if (i == 0) { bar[0] = 0u; bar[1] = 0u; }
}

// ---------------------------------------------------------------------------
// GRU recurrence: persistent 16-workgroup kernel, one barrier per timestep.
// Each workgroup owns 32 columns of H. h_prev is staged into LDS as f16 each
// step (cooperative convert), so A-fragments are ds_load_b128, cvt-free.
// Wave w: row-tile rt = w&3 (B/16 = 4), col-tile ct = w>>2; 3 gate accums.
// Epilogue operands (xg slices, h_prev f32) are loaded BEFORE the k-loop so
// their latency hides under the 48 WMMAs on the serial critical path.
// ---------------------------------------------------------------------------
__global__ __launch_bounds__(256) void gru_rec_kernel(
    const float* __restrict__ xg, const _Float16* __restrict__ WhhP,
    const float* __restrict__ bhh, float* __restrict__ hbuf,
    float* __restrict__ Y, _Float16* __restrict__ Yh,
    unsigned* __restrict__ bar) {
  __shared__ _Float16 hs[BATCH * HID];  // 64 KB f16 state tile

  const int wave = threadIdx.x >> 5;
  const int lane = threadIdx.x & 31;
  const int c0   = blockIdx.x * 32;       // owned column base (H slice)
  const int rt   = wave & 3;              // row tile
  const int ct   = wave >> 2;             // col tile within the 32-col slice
  const int m0   = rt * 16;
  const int nc   = c0 + ct * 16;
  const int nl   = lane & 15;
  const int mb   = m0 + ((lane >> 4) << 3);
  const int col  = nc + nl;
  const float br = bhh[col];
  const float bz = bhh[HID + col];
  const float bn = bhh[2 * HID + col];

  for (int t = 0; t < T_STEPS; ++t) {
    const float* hprev = hbuf + (size_t)(t & 1) * BATCH * HID;
    float*       hnext = hbuf + (size_t)((t + 1) & 1) * BATCH * HID;
    const float* xgt   = xg + (size_t)t * BATCH * G3H;

    // Stage h_prev (f32, global/L2) -> hs (f16, LDS), fully coalesced.
#pragma unroll
    for (int c = 0; c < (BATCH * HID) / (256 * 4); ++c) {
      int idx = (c * 256 + (int)threadIdx.x) * 4;
      v4f v = *(const v4f*)(hprev + idx);
      v4h h4;
#pragma unroll
      for (int j = 0; j < 4; ++j) h4[j] = (_Float16)v[j];
      *(v4h*)(hs + idx) = h4;
    }
    __syncthreads();

    // Issue epilogue loads early: they complete under the WMMA phase.
    float xr[8], xz[8], xn[8], hp[8];
#pragma unroll
    for (int v = 0; v < 8; ++v) {
      const int row = mb + v;
      xr[v] = xgt[(size_t)row * G3H + col];
      xz[v] = xgt[(size_t)row * G3H + HID + col];
      xn[v] = xgt[(size_t)row * G3H + 2 * HID + col];
      hp[v] = hprev[(size_t)row * HID + col];
    }

    v8f acc[3];
#pragma unroll
    for (int g = 0; g < 3; ++g)
#pragma unroll
      for (int j = 0; j < 8; ++j) acc[g][j] = 0.0f;

    v16h a_cur = load_a_frag_f16(hs, HID, m0, 0);
    v16h b_cur[3];
#pragma unroll
    for (int g = 0; g < 3; ++g) b_cur[g] = load_b_frag(WhhP, G3H >> 4, g * HID + nc, 0);

    for (int k0 = 0; k0 < HID; k0 += 32) {
      v16h a_nxt = a_cur;
      v16h b_nxt[3];
      const int kn = k0 + 32;
      if (kn < HID) {
        a_nxt = load_a_frag_f16(hs, HID, m0, kn);
#pragma unroll
        for (int g = 0; g < 3; ++g) b_nxt[g] = load_b_frag(WhhP, G3H >> 4, g * HID + nc, kn);
      }
#pragma unroll
      for (int g = 0; g < 3; ++g)
        acc[g] = __builtin_amdgcn_wmma_f32_16x16x32_f16(false, a_cur, false, b_cur[g], (short)0, acc[g], false, false);
      a_cur = a_nxt;
#pragma unroll
      for (int g = 0; g < 3; ++g) b_cur[g] = (kn < HID) ? b_nxt[g] : b_cur[g];
    }

    // Gate math directly on the C-fragment layout.
#pragma unroll
    for (int v = 0; v < 8; ++v) {
      const int row = mb + v;
      const size_t ri = (size_t)row * HID + col;
      float r  = sigmoidf_(xr[v] + acc[0][v] + br);
      float z  = sigmoidf_(xz[v] + acc[1][v] + bz);
      float n  = tanhf(xn[v] + r * (acc[2][v] + bn));
      float hv = (1.0f - z) * n + z * hp[v];
      hnext[ri] = hv;
      Y[(size_t)t * BATCH * HID + ri] = hv;
      Yh[(size_t)t * BATCH * HID + ri] = (_Float16)hv;
    }

    // Hint-prefetch next timestep's xg slab (lowers to global_prefetch_b8).
    if (t + 1 < T_STEPS) {
      const float* nx = xg + (size_t)(t + 1) * BATCH * G3H + (size_t)threadIdx.x * 64;
      __builtin_prefetch(nx, 0, 1);
    }

    // Grid-wide barrier between timesteps (16 co-resident blocks).
    __syncthreads();
    if (threadIdx.x == 0) {
      __threadfence();
      unsigned a = atomicAdd(&bar[0], 1u);
      if (a == REC_WGS - 1) {
        atomicExch(&bar[0], 0u);
        __threadfence();
        atomicExch(&bar[1], (unsigned)(t + 1));
      } else {
        while (atomicAdd(&bar[1], 0u) < (unsigned)(t + 1)) {
          __builtin_amdgcn_s_sleep(2);
        }
      }
    }
    __syncthreads();
  }
}

// ---------------------------------------------------------------------------
// Host driver
// ---------------------------------------------------------------------------
extern "C" void kernel_launch(void* const* d_in, const int* in_sizes, int n_in,
                              void* d_out, int out_size, void* d_ws, size_t ws_size,
                              hipStream_t stream) {
  (void)in_sizes; (void)n_in; (void)out_size; (void)ws_size;

  const float* x      = (const float*)d_in[0];
  const float* g1Wih  = (const float*)d_in[1];
  const float* g1Whh  = (const float*)d_in[2];
  const float* g1bih  = (const float*)d_in[3];
  const float* g1bhh  = (const float*)d_in[4];
  const float* linW   = (const float*)d_in[5];
  const float* linb   = (const float*)d_in[6];
  const float* g2Wih  = (const float*)d_in[7];
  const float* g2Whh  = (const float*)d_in[8];
  const float* g2bih  = (const float*)d_in[9];
  const float* g2bhh  = (const float*)d_in[10];
  float* out          = (float*)d_out;

  // Workspace carve-out
  uint8_t* ws = (uint8_t*)d_ws;
  size_t off = 0;
  auto carve = [&](size_t bytes) -> void* {
    void* p = ws + off;
    off += (bytes + 255) & ~(size_t)255;
    return p;
  };
  const size_t WSZ = (size_t)G3H * HID;        // elements per GRU weight matrix
  const size_t ACT = (size_t)T_STEPS * BATCH * HID;
  _Float16* pWih  = (_Float16*)carve(4 * WSZ * sizeof(_Float16));
  _Float16* pWhh  = (_Float16*)carve(4 * WSZ * sizeof(_Float16));
  _Float16* pLin  = (_Float16*)carve((size_t)HID * HID * sizeof(_Float16));
  float*    xg    = (float*)carve((size_t)T_STEPS * BATCH * G3H * sizeof(float));
  float*    actA  = (float*)carve(ACT * sizeof(float));
  float*    actB  = (float*)carve(ACT * sizeof(float));
  _Float16* xh    = (_Float16*)carve(ACT * sizeof(_Float16));
  _Float16* actAh = (_Float16*)carve(ACT * sizeof(_Float16));
  _Float16* actBh = (_Float16*)carve(ACT * sizeof(_Float16));
  float*    hbuf  = (float*)carve(2 * (size_t)BATCH * HID * sizeof(float));
  unsigned* bar   = (unsigned*)carve(256);

  auto pack = [&](const float* W, _Float16* P, int N) {
    size_t total = (size_t)N * HID;
    pack_w_kernel<<<dim3((unsigned)((total + 255) / 256)), dim3(256), 0, stream>>>(W, P, N, HID);
  };
  for (int l = 0; l < 2; ++l) {
    pack(g1Wih + (size_t)l * WSZ, pWih + (size_t)l * WSZ, G3H);
    pack(g1Whh + (size_t)l * WSZ, pWhh + (size_t)l * WSZ, G3H);
    pack(g2Wih + (size_t)l * WSZ, pWih + (size_t)(2 + l) * WSZ, G3H);
    pack(g2Whh + (size_t)l * WSZ, pWhh + (size_t)(2 + l) * WSZ, G3H);
  }
  pack(linW, pLin, HID);
  cvt_f16_kernel<<<dim3((unsigned)((ACT / 4 + 255) / 256)), dim3(256), 0, stream>>>(x, xh, (int)(ACT / 4));

  const int TB = T_STEPS * BATCH;  // 16384 rows

  auto run_layer = [&](const _Float16* inActH, const _Float16* WihP, const _Float16* WhhP,
                       const float* bih, const float* bhh, float* outY, _Float16* outYh) {
    gemm_bias_kernel<<<dim3(TB / 64, G3H / 128), dim3(256), 0, stream>>>(inActH, WihP, bih, xg, TB, G3H);
    init_state_kernel<<<dim3((2 * BATCH * HID + 255) / 256), dim3(256), 0, stream>>>(hbuf, bar);
    gru_rec_kernel<<<dim3(REC_WGS), dim3(256), 0, stream>>>(xg, WhhP, bhh, hbuf, outY, outYh, bar);
  };

  // Stack 1
  run_layer(xh,    pWih + 0 * WSZ, pWhh + 0 * WSZ, g1bih,       g1bhh,       actA, actAh);
  run_layer(actAh, pWih + 1 * WSZ, pWhh + 1 * WSZ, g1bih + G3H, g1bhh + G3H, actB, actBh);
  // Gated residual mix: reads y (f16 for GEMM, f32 for mix), x (f32 residual)
  gate_mix_kernel<<<dim3(TB / 64, HID / 128), dim3(256), 0, stream>>>(
      actBh, actB, pLin, linb, x, actA, actAh, TB);
  // Stack 2
  run_layer(actAh, pWih + 2 * WSZ, pWhh + 2 * WSZ, g2bih,       g2bhh,       actB, actBh);
  run_layer(actBh, pWih + 3 * WSZ, pWhh + 3 * WSZ, g2bih + G3H, g2bhh + G3H, out,  actAh);
}
